// Sc_encoder_63806034149592
// MI455X (gfx1250) — matrix operand order repaired and checked by reading, not compiled, not run on hardware
//
#include <hip/hip_runtime.h>
#include <cstdint>
#include <cstddef>

// ---------------------------------------------------------------------------
// MI455X (gfx1250) wave32 WMMA implementation of the 2-relation GAT + semantic
// attention reference.  All dense GEMMs run on v_wmma_f32_16x16x32_f16 with
// f32 accumulation.  feat_src matrices are kept in f16 so the random edge
// gather working set (102 MB for src0) stays resident in the 192 MB L2.
// feat_dst is never materialized (only er = feat_dst @ a_r, fused in the GEMM
// epilogue).  tanh+mean over W_fc is fused into the GEMM as per-block column
// sums, reduced in fixed order (deterministic, no float atomics).
// All bulk global traffic is 16B-per-lane coalesced (float4 / b128).
// ---------------------------------------------------------------------------

typedef __attribute__((ext_vector_type(16))) _Float16 v16h;
typedef __attribute__((ext_vector_type(8)))  _Float16 v8h;
typedef __attribute__((ext_vector_type(4)))  _Float16 v4h;
typedef __attribute__((ext_vector_type(8)))  float    v8f;

#define DD 128
#define ROWS_PER_BLOCK 64     // 4 waves x 16 rows
#define THREADS 128
#define XSTRIDE 136           // 128 + 8 f16 padding (272 B rows, 16B aligned)

__device__ __forceinline__ float reduce16_add(float v) {
  v += __shfl_xor(v, 1, 32);
  v += __shfl_xor(v, 2, 32);
  v += __shfl_xor(v, 4, 32);
  v += __shfl_xor(v, 8, 32);
  return v;
}

// Y[N,128] = X[N,128] @ B[128,128]   (B = W, or W^T if TRANSB)
//   STOREY : write Y as f16 (LDS-bounced, coalesced b128 stores)
//   EL     : elv[row] = dot(Y[row,:], avec)
//   TANHSUM: colsum[block, n] = sum_rows tanh(Y[row,n] + bias[n])  (valid rows)
template<bool STOREY, bool EL, bool TANHSUM, bool TRANSB>
__global__ __launch_bounds__(THREADS)
void gemm128_wmma(const float* __restrict__ X,
                  const float* __restrict__ W,
                  const float* __restrict__ avec,
                  const float* __restrict__ bias,
                  _Float16* __restrict__ Yh,
                  float* __restrict__ elv,
                  float* __restrict__ colsum,
                  int N)
{
  // B fragments pre-packed per (kstep, ntile, lane): 16 contiguous f16 each.
  __shared__ alignas(32) _Float16 sWpk[4 * 8 * 32 * 16];        // 32 KB
  __shared__ alignas(16) _Float16 sX[ROWS_PER_BLOCK * XSTRIDE]; // 17 KB
  __shared__ float sCol[4 * DD];                                // 2 KB

  const int tid  = threadIdx.x;
  const int wave = tid >> 5;
  const int l    = tid & 31;
  const int sel  = l >> 4;       // half-wave selector
  const int ln   = l & 15;
  const int rowbase = blockIdx.x * ROWS_PER_BLOCK;

  // --- pack W into WMMA B-fragment layout (coalesced float4 loads) ---------
  // B frag element e of lane lc:  B[k][n], k = ks*32 + (lc>>4)*16 + e,
  //                               n = nt*16 + (lc&15)
  {
    const float4* W4 = (const float4*)W;
    #pragma unroll
    for (int i = 0; i < 32; ++i) {
      int fid = tid + i * THREADS;        // 0..4095 float4 chunks of W
      int a   = fid >> 5;                 // source row of the load
      int b4  = (fid & 31) * 4;           // source col start
      float4 wv = W4[fid];                // coalesced
      #pragma unroll
      for (int j = 0; j < 4; ++j) {
        int k, n;
        if (TRANSB) { n = a; k = b4 + j; }   // B[k][n] = W[n][k]
        else        { k = a; n = b4 + j; }   // B[k][n] = W[k][n]
        float w = (&wv.x)[j];
        int ks = k >> 5;
        int e  = k & 15;
        int lc = ((k >> 4) & 1) * 16 + (n & 15);
        int nt = n >> 4;
        int cid = (ks * 8 + nt) * 32 + lc;
        sWpk[cid * 16 + e] = (_Float16)w;
      }
    }
  }
  // --- stage X tile in LDS (coalesced float4 loads, f32 -> f16) ------------
  #pragma unroll
  for (int i = 0; i < (ROWS_PER_BLOCK * DD) / (4 * THREADS); ++i) {
    int fid = tid + i * THREADS;          // 0..2047 float4 chunks
    int r   = fid >> 5;                   // 32 float4 per row
    int c4  = (fid & 31) * 4;
    int gr  = rowbase + r;
    float4 xv = make_float4(0.f, 0.f, 0.f, 0.f);
    if (gr < N) xv = ((const float4*)X)[(size_t)gr * (DD / 4) + (fid & 31)];
    v4h h;
    h[0] = (_Float16)xv.x; h[1] = (_Float16)xv.y;
    h[2] = (_Float16)xv.z; h[3] = (_Float16)xv.w;
    *(v4h*)&sX[r * XSTRIDE + c4] = h;
  }
  __syncthreads();

  v8f acc[8] = {};

  const int rowloc = wave * 16 + ln;
  #pragma unroll
  for (int ks = 0; ks < 4; ++ks) {
    // A frag (16x32 f16): lane (sel,ln) holds row ln, K = ks*32 + sel*16 + {0..15}
    // (contiguous order, matching the B packing)
    const _Float16* xr = &sX[rowloc * XSTRIDE + ks * 32 + sel * 16];
    v8h alo = *(const v8h*)(xr);
    v8h ahi = *(const v8h*)(xr + 8);
    v16h a = __builtin_shufflevector(alo, ahi,
                0,1,2,3,4,5,6,7,8,9,10,11,12,13,14,15);
    #pragma unroll
    for (int nt = 0; nt < 8; ++nt) {
      v16h b = *(const v16h*)&sWpk[(size_t)(((ks * 8 + nt) * 32) + l) * 16];
      acc[nt] = __builtin_amdgcn_wmma_f32_16x16x32_f16(
          false, a, false, b, (short)0, acc[nt], false, false);
    }
  }

  // C/D layout: VGPR r of lane l -> row (sel*8 + r), col (l&15)  per 16x16 tile
  const int rbase = rowbase + wave * 16 + sel * 8;

  if constexpr (EL) {
    float av[8];
    #pragma unroll
    for (int nt = 0; nt < 8; ++nt) av[nt] = avec[nt * 16 + ln];
    #pragma unroll
    for (int r = 0; r < 8; ++r) {
      float p = 0.f;
      #pragma unroll
      for (int nt = 0; nt < 8; ++nt) p += acc[nt][r] * av[nt];
      p = reduce16_add(p);               // sum over 16 columns held per lane grp
      int row = rbase + r;
      if (ln == 0 && row < N) elv[row] = p;
    }
  }

  if constexpr (TANHSUM) {
    float bv[8];
    #pragma unroll
    for (int nt = 0; nt < 8; ++nt) bv[nt] = bias[nt * 16 + ln];
    #pragma unroll
    for (int nt = 0; nt < 8; ++nt) {
      float s = 0.f;
      #pragma unroll
      for (int r = 0; r < 8; ++r) {
        int row = rbase + r;
        if (row < N) s += tanhf(acc[nt][r] + bv[nt]);
      }
      s += __shfl_xor(s, 16, 32);        // rows 0-7 + rows 8-15, same column
      if (sel == 0) sCol[wave * DD + nt * 16 + ln] = s;
    }
    __syncthreads();
    if (tid < DD) {
      float t = 0.f;
      #pragma unroll
      for (int w = 0; w < 4; ++w) t += sCol[w * DD + tid];
      colsum[(size_t)blockIdx.x * DD + tid] = t;
    }
  }

  if constexpr (STOREY) {
    // Each wave owns rows [wave*16, wave*16+16) of sX exclusively -> safe to
    // overwrite its own rows, then one barrier before the coalesced drain.
    #pragma unroll
    for (int nt = 0; nt < 8; ++nt) {
      #pragma unroll
      for (int r = 0; r < 8; ++r)
        sX[(wave * 16 + sel * 8 + r) * XSTRIDE + nt * 16 + ln] =
            (_Float16)acc[nt][r];
    }
    __syncthreads();
    #pragma unroll
    for (int i = 0; i < 8; ++i) {
      int chunk = tid + i * THREADS;     // 0..1023 : 64 rows x 16 b128 chunks
      int r = chunk >> 4;
      int c = (chunk & 15) * 8;
      int row = rowbase + r;
      if (row < N) {
        v8h v = *(const v8h*)&sX[r * XSTRIDE + c];
        *(v8h*)&Yh[(size_t)row * DD + c] = v;        // global_store_b128
      }
    }
  }
}

// Per-dst-node edge softmax + weighted gather of f16 source features.
// One wave per dst node; each lane owns 4 contiguous columns -> b64 gather
// loads (256 B per wave per edge) and float4 output stores.
__global__ __launch_bounds__(128)
void gat_gather(const _Float16* __restrict__ feat,
                const float* __restrict__ el,
                const float* __restrict__ er,
                const float* __restrict__ bvec,
                const int* __restrict__ idx,
                float* __restrict__ eout,
                int K, int ND)
{
  const int wave = threadIdx.x >> 5;
  const int lane = threadIdx.x & 31;
  const int n = blockIdx.x * 4 + wave;
  if (n >= ND) return;
  const int c0 = lane * 4;

  int   ids[8];
  float ee[8];
  const float ern = er[n];              // wave-uniform -> scalar load
  float m = -1e30f;
  for (int k = 0; k < K; ++k) {
    int id = idx[(size_t)n * K + k];    // wave-uniform -> scalar load
    ids[k] = id;
    float x = el[id] + ern;
    x = (x >= 0.f) ? x : 0.2f * x;      // leaky_relu(0.2)
    ee[k] = x;
    m = fmaxf(m, x);
  }
  float denom = 0.f;
  for (int k = 0; k < K; ++k) { ee[k] = __expf(ee[k] - m); denom += ee[k]; }
  const float inv = 1.0f / denom;

  float a0 = bvec[c0 + 0], a1 = bvec[c0 + 1];
  float a2 = bvec[c0 + 2], a3 = bvec[c0 + 3];
  for (int k = 0; k < K; ++k) {
    const float w = ee[k] * inv;
    v4h f = *(const v4h*)&feat[(size_t)ids[k] * DD + c0];   // b64 gather
    a0 += w * (float)f[0];
    a1 += w * (float)f[1];
    a2 += w * (float)f[2];
    a3 += w * (float)f[3];
  }
  float4 o; o.x = a0; o.y = a1; o.z = a2; o.w = a3;
  *(float4*)&eout[(size_t)n * DD + c0] = o;                 // b128 store
}

// Deterministic reduction of per-block column sums -> sp0/sp1 -> beta softmax.
__global__ __launch_bounds__(128)
void beta_kernel(const float* __restrict__ cs0,
                 const float* __restrict__ cs1,
                 const float* __restrict__ att,
                 int nblk, float invN, float* __restrict__ beta)
{
  __shared__ float red[DD];
  const int d = threadIdx.x;
  float s0 = 0.f, s1 = 0.f;
  for (int b = 0; b < nblk; ++b) {
    s0 += cs0[(size_t)b * DD + d];
    s1 += cs1[(size_t)b * DD + d];
  }
  const float a = att[d];
  const float v0 = a * s0 * invN;
  const float v1 = a * s1 * invN;
  red[d] = v0; __syncthreads();
  for (int off = 64; off > 0; off >>= 1) {
    if (d < off) red[d] += red[d + off];
    __syncthreads();
  }
  const float S0 = red[0]; __syncthreads();
  red[d] = v1; __syncthreads();
  for (int off = 64; off > 0; off >>= 1) {
    if (d < off) red[d] += red[d + off];
    __syncthreads();
  }
  if (d == 0) {
    const float S1 = red[0];
    const float mm = fmaxf(S0, S1);
    const float z0 = __expf(S0 - mm), z1 = __expf(S1 - mm);
    const float zi = 1.f / (z0 + z1);
    beta[0] = z0 * zi;
    beta[1] = z1 * zi;
  }
}

__global__ void combine_kernel(const float4* __restrict__ e0,
                               const float4* __restrict__ e1,
                               const float* __restrict__ beta,
                               float4* __restrict__ out, size_t total4)
{
  size_t i = (size_t)blockIdx.x * blockDim.x + threadIdx.x;
  if (i < total4) {
    const float b0 = beta[0], b1 = beta[1];
    float4 x = e0[i], y = e1[i], o;
    o.x = b0 * x.x + b1 * y.x;
    o.y = b0 * x.y + b1 * y.y;
    o.z = b0 * x.z + b1 * y.z;
    o.w = b0 * x.w + b1 * y.w;
    out[i] = o;
  }
}

extern "C" void kernel_launch(void* const* d_in, const int* in_sizes, int n_in,
                              void* d_out, int out_size, void* d_ws, size_t ws_size,
                              hipStream_t stream)
{
  const float* h_src0 = (const float*)d_in[0];
  const float* h_src1 = (const float*)d_in[1];
  const float* h_dst  = (const float*)d_in[2];
  const int*   idx0   = (const int*)d_in[3];
  const int*   idx1   = (const int*)d_in[4];
  const float* W_src0 = (const float*)d_in[5];
  const float* W_dst0 = (const float*)d_in[6];
  const float* a_l0   = (const float*)d_in[7];
  const float* a_r0   = (const float*)d_in[8];
  const float* b0     = (const float*)d_in[9];
  const float* W_src1 = (const float*)d_in[10];
  const float* W_dst1 = (const float*)d_in[11];
  const float* a_l1   = (const float*)d_in[12];
  const float* a_r1   = (const float*)d_in[13];
  const float* b1     = (const float*)d_in[14];
  const float* W_fc   = (const float*)d_in[15];
  const float* b_fc   = (const float*)d_in[16];
  const float* att    = (const float*)d_in[17];

  const int N0 = in_sizes[0] / DD;   // 400000
  const int N1 = in_sizes[1] / DD;   // 10000
  const int ND = in_sizes[2] / DD;   // 100000
  const int K0 = in_sizes[3] / ND;   // 8
  const int K1 = in_sizes[4] / ND;   // 4

  // workspace carve (256B aligned)
  uint8_t* base = (uint8_t*)d_ws;
  size_t off = 0;
  auto carve = [&](size_t bytes) -> void* {
    void* r = base + off;
    off = (off + bytes + 255) & ~(size_t)255;
    return r;
  };
  const int nblkD = (ND + ROWS_PER_BLOCK - 1) / ROWS_PER_BLOCK;
  _Float16* feat0 = (_Float16*)carve((size_t)N0 * DD * sizeof(_Float16)); // 102 MB (L2-resident)
  _Float16* feat1 = (_Float16*)carve((size_t)N1 * DD * sizeof(_Float16));
  float* el0 = (float*)carve((size_t)N0 * sizeof(float));
  float* el1 = (float*)carve((size_t)N1 * sizeof(float));
  float* er0 = (float*)carve((size_t)ND * sizeof(float));
  float* er1 = (float*)carve((size_t)ND * sizeof(float));
  float* e0  = (float*)carve((size_t)ND * DD * sizeof(float));
  float* e1  = (float*)carve((size_t)ND * DD * sizeof(float));
  float* cs0 = (float*)carve((size_t)nblkD * DD * sizeof(float));
  float* cs1 = (float*)carve((size_t)nblkD * DD * sizeof(float));
  float* betab = (float*)carve(2 * sizeof(float));

  const int nblk0 = (N0 + ROWS_PER_BLOCK - 1) / ROWS_PER_BLOCK;
  const int nblk1 = (N1 + ROWS_PER_BLOCK - 1) / ROWS_PER_BLOCK;

  // feat_src + el (stored f16);  feat_dst is never stored — only er.
  gemm128_wmma<true,  true,  false, false><<<nblk0, THREADS, 0, stream>>>(
      h_src0, W_src0, a_l0, nullptr, feat0, el0, nullptr, N0);
  gemm128_wmma<false, true,  false, false><<<nblkD, THREADS, 0, stream>>>(
      h_dst,  W_dst0, a_r0, nullptr, nullptr, er0, nullptr, ND);
  gemm128_wmma<true,  true,  false, false><<<nblk1, THREADS, 0, stream>>>(
      h_src1, W_src1, a_l1, nullptr, feat1, el1, nullptr, N1);
  gemm128_wmma<false, true,  false, false><<<nblkD, THREADS, 0, stream>>>(
      h_dst,  W_dst1, a_r1, nullptr, nullptr, er1, nullptr, ND);

  // edge softmax + weighted gather (f16 feats served from L2)
  const int gblk = (ND + 3) / 4;
  gat_gather<<<gblk, 128, 0, stream>>>(feat0, el0, er0, b0, idx0, e0, K0, ND);
  gat_gather<<<gblk, 128, 0, stream>>>(feat1, el1, er1, b1, idx1, e1, K1, ND);

  // fused e @ W_fc^T + b_fc -> tanh -> per-block column sums
  gemm128_wmma<false, false, true,  true ><<<nblkD, THREADS, 0, stream>>>(
      e0, W_fc, nullptr, b_fc, nullptr, nullptr, cs0, ND);
  gemm128_wmma<false, false, true,  true ><<<nblkD, THREADS, 0, stream>>>(
      e1, W_fc, nullptr, b_fc, nullptr, nullptr, cs1, ND);

  beta_kernel<<<1, DD, 0, stream>>>(cs0, cs1, att, nblkD, 1.0f / (float)ND, betab);

  const size_t total4 = (size_t)ND * DD / 4;
  const int cb = (int)((total4 + 255) / 256);
  combine_kernel<<<cb, 256, 0, stream>>>((const float4*)e0, (const float4*)e1,
                                         betab, (float4*)d_out, total4);
}
